// SmallGPT_48533130445066
// MI455X (gfx1250) — matrix operand order repaired and checked by reading, not compiled, not run on hardware
//
#include <hip/hip_runtime.h>
#include <hip/hip_bf16.h>
#include <math.h>

// ---------------- model constants ----------------
#define B_      2
#define S_      1024
#define D_      1024
#define L_      8
#define HQ_     16
#define HKV_    4
#define HD_     64
#define H_      2752
#define V_      32000
#define WINDOW_ 256
#define NEGF    (-1.0e30f)

typedef _Float16 half_t;
typedef __attribute__((ext_vector_type(16))) _Float16 v16h;
typedef __attribute__((ext_vector_type(8)))  float    v8f;

union V16 { uint4 u[2]; v16h v; };

__device__ __forceinline__ v8f wmma32(v16h a, v16h b, v8f c) {
  // D = A(16x32 f16) * B(32x16 f16) + C(16x16 f32)
  return __builtin_amdgcn_wmma_f32_16x16x32_f16(false, a, false, b, (short)0, c,
                                                false, false);
}

// Async direct global->LDS copy of 32 bytes (two b128 ops; the ISA adds the
// instruction offset to BOTH the global and the LDS address). Tracked by
// ASYNCcnt. Generic->LDS address: low 32 bits of the flat address are the
// LDS offset (aperture scheme), so a truncating cast is sufficient.
__device__ __forceinline__ void async_b128x2(unsigned lds, const half_t* g) {
  asm volatile(
      "global_load_async_to_lds_b128 %0, %1, off\n\t"
      "global_load_async_to_lds_b128 %0, %1, off offset:16"
      :: "v"(lds), "v"(g) : "memory");
}
__device__ __forceinline__ void wait_async0() {
  asm volatile("s_wait_asynccnt 0x0" ::: "memory");
}

// ---------------- embedding gather ----------------
__global__ void embed_gather(const int* __restrict__ idx,
                             const float* __restrict__ emb,
                             float* __restrict__ x, int n) {
  int i = blockIdx.x * blockDim.x + threadIdx.x;
  if (i >= n) return;
  int m = i >> 10;          // D_ == 1024
  int d = i & 1023;
  x[i] = emb[(size_t)idx[m] * D_ + d];
}

// ---------------- RMSNorm -> f16 ----------------
__global__ __launch_bounds__(256)
void rmsnorm_f16(const float* __restrict__ x, const float* __restrict__ g,
                 half_t* __restrict__ out) {
  __shared__ float red[256];
  const int row = blockIdx.x;
  const float* xr = x + (size_t)row * D_;
  float ss = 0.0f;
  for (int i = threadIdx.x; i < D_; i += 256) { float v = xr[i]; ss += v * v; }
  red[threadIdx.x] = ss;
  __syncthreads();
  for (int off = 128; off > 0; off >>= 1) {
    if (threadIdx.x < off) red[threadIdx.x] += red[threadIdx.x + off];
    __syncthreads();
  }
  const float rs = rsqrtf(red[0] * (1.0f / D_) + 1e-6f);
  half_t* orow = out + (size_t)row * D_;
  for (int i = threadIdx.x; i < D_; i += 256)
    orow[i] = (half_t)(xr[i] * rs * g[i]);
}

// ------- RoPE (optional) + repack [B,S,nh*64] f32 -> [B,nh,S,64] f16 -------
__global__ void rope_pack(const float* __restrict__ in, half_t* __restrict__ out,
                          int nh, int doRope) {
  int i = blockIdx.x * blockDim.x + threadIdx.x;
  int total = B_ * S_ * nh * (HD_ / 2);
  if (i >= total) return;
  int j = i & 31;           // pair index, HD_/2 == 32
  int t = i >> 5;
  int hh = t % nh; t /= nh;
  int s  = t % S_;
  int b  = t / S_;
  const float* src = in + ((size_t)(b * S_ + s) * nh + hh) * HD_ + 2 * j;
  float x0 = src[0], x1 = src[1];
  half_t* dst = out + (((size_t)(b * nh + hh) * S_ + s) * HD_) + 2 * j;
  if (doRope) {
    float inv = __powf(10000.0f, -(float)(2 * j) * (1.0f / 64.0f));
    float th = (float)s * inv;
    float sn, cs;
    __sincosf(th, &sn, &cs);
    dst[0] = (half_t)(x0 * cs - x1 * sn);
    dst[1] = (half_t)(x0 * sn + x1 * cs);
  } else {
    dst[0] = (half_t)x0;
    dst[1] = (half_t)x1;
  }
}

// ---------------- SwiGLU -> f16 ----------------
__global__ void swiglu_f16(const float* __restrict__ u, const float* __restrict__ g,
                           half_t* __restrict__ out, int n) {
  int i = blockIdx.x * blockDim.x + threadIdx.x;
  if (i >= n) return;
  float uv = u[i];
  float s = uv / (1.0f + __expf(-uv));
  out[i] = (half_t)(s * g[i]);
}

// ---------------- generic WMMA GEMM (async double-buffered) ----------------
// C[M,N] f32 = A[M,K] f16 (row-major) * W (f32) [+ R]
// bTrans==0: W is [K,N] row-major.  bTrans==1: W is [N,K] row-major (W^T used).
// Tile: BM=128, BN=64, BK=32. 256 threads = 8 waves; wave grid 4(M)x2(N),
// each wave computes 32x32 via 2x2 v_wmma_f32_16x16x32_f16 tiles.
// A tile is staged with global_load_async_to_lds_b128 (ASYNCcnt), double
// buffered so tile i+1 streams in while tile i feeds the WMMAs.
__global__ __launch_bounds__(256)
void gemm_wmma(const half_t* __restrict__ A, const float* __restrict__ W,
               const float* __restrict__ R, float* __restrict__ C,
               int M, int N, int K, int bTrans) {
  __shared__ __align__(16) half_t sA[2][128 * 32];  // row-major [row][k]
  __shared__ __align__(16) half_t sB[2][64 * 32];   // col-major [col][k]
  const int tid  = threadIdx.x;
  const int lane = tid & 31;
  const int lh   = lane & 15;
  const int sel  = lane >> 4;        // 0: lanes 0-15, 1: lanes 16-31
  const int wave = tid >> 5;
  const int wm = (wave & 3) * 32;    // wave M offset within tile
  const int wn = (wave >> 2) * 32;   // wave N offset within tile
  const int n0 = blockIdx.x * 64;
  const int m0 = blockIdx.y * 128;

  v8f acc[2][2];
#pragma unroll
  for (int a = 0; a < 2; ++a)
#pragma unroll
    for (int b = 0; b < 2; ++b)
#pragma unroll
      for (int i = 0; i < 8; ++i) acc[a][b][i] = 0.0f;

  const int arow = tid >> 1, akoff = (tid & 1) * 16;

  auto stage = [&](int buf, int kb) {
    // A tile: pure f16 copy, async direct to LDS
    async_b128x2((unsigned)(uintptr_t)(&sA[buf][arow * 32 + akoff]),
                 A + (size_t)(m0 + arow) * K + kb + akoff);
    // B tile: f32 -> f16 convert through VGPRs, store col-major
    if (!bTrans) {
      int k = tid >> 3, j0 = (tid & 7) * 8;
      const float* src = W + (size_t)(kb + k) * N + n0 + j0;
      if (kb + 32 < K) __builtin_prefetch(src + (size_t)32 * N, 0, 0);
#pragma unroll
      for (int j = 0; j < 8; ++j) sB[buf][(j0 + j) * 32 + k] = (half_t)src[j];
    } else {
      int col = tid >> 2, c0 = (tid & 3) * 8;
      const float* src = W + (size_t)(n0 + col) * K + kb + c0;
      if (kb + 32 < K) __builtin_prefetch(src + 32, 0, 0);
#pragma unroll
      for (int j = 0; j < 8; ++j) sB[buf][col * 32 + c0 + j] = (half_t)src[j];
    }
  };

  stage(0, 0);
  wait_async0();
  __syncthreads();

  int buf = 0;
  for (int kb = 0; kb < K; kb += 32, buf ^= 1) {
    if (kb + 32 < K) stage(buf ^ 1, kb + 32);

    V16 af[2], bf[2];
#pragma unroll
    for (int mt = 0; mt < 2; ++mt) {
      // A 16x32 frag: lanes<16 hold K 0..7 & 16..23, lanes>=16 hold 8..15 & 24..31
      const uint4* pa = (const uint4*)(&sA[buf][(wm + mt * 16 + lh) * 32]);
      af[mt].u[0] = pa[sel];
      af[mt].u[1] = pa[2 + sel];
    }
#pragma unroll
    for (int nt = 0; nt < 2; ++nt) {
      // B 32x16 frag: lane = column, lanes<16 hold K 0..7 & 16..23, lanes>=16 hold 8..15 & 24..31
      const uint4* pb = (const uint4*)(&sB[buf][(wn + nt * 16 + lh) * 32]);
      bf[nt].u[0] = pb[sel];
      bf[nt].u[1] = pb[2 + sel];
    }
#pragma unroll
    for (int mt = 0; mt < 2; ++mt)
#pragma unroll
      for (int nt = 0; nt < 2; ++nt)
        acc[mt][nt] = wmma32(af[mt].v, bf[nt].v, acc[mt][nt]);

    wait_async0();     // next-buffer async copies landed in LDS
    __syncthreads();   // everyone done reading buf; next buf visible to all
  }

  // epilogue: C layout — VGPR i: lanes<16 M=i, lanes>=16 M=8+i; N=lane&15
#pragma unroll
  for (int mt = 0; mt < 2; ++mt)
#pragma unroll
    for (int nt = 0; nt < 2; ++nt) {
      int col = n0 + wn + nt * 16 + lh;
#pragma unroll
      for (int i = 0; i < 8; ++i) {
        int row = m0 + wm + mt * 16 + i + sel * 8;
        size_t o = (size_t)row * N + col;
        float v = acc[mt][nt][i];
        if (R) v += R[o];
        C[o] = v;
      }
    }
}

// ---------------- flash attention (wave32 WMMA) ----------------
// Q:[B,HQ,S,64] f16, K/V:[B,HKV,S,64] f16, O:[B,S,HQ*64] f16
// Block = 128 thr (4 waves), one (b,h), 64 query rows (16 per wave).
// Computes S^T = K*Q^T so a query column lives in a lane pair (L, L^16):
// softmax reduce = 16 in-lane values + one xor-16 shuffle. Then O^T = V^T*P.
// K tile staged async direct-to-LDS; V tile transposed through VGPRs.
__global__ __launch_bounds__(128)
void flash_attn(const half_t* __restrict__ Q, const half_t* __restrict__ Kc,
                const half_t* __restrict__ Vc, half_t* __restrict__ O,
                int useGlobal) {
  __shared__ __align__(16) half_t sK[32 * 64];   // [key][d]
  __shared__ __align__(16) half_t sVt[64 * 32];  // [d][key]
  const int tid  = threadIdx.x;
  const int lane = tid & 31;
  const int lh   = lane & 15;
  const int sel  = lane >> 4;
  const int wave = tid >> 5;
  const int bh = blockIdx.y;
  const int b = bh / HQ_, h = bh % HQ_, kvh = h >> 2;   // GQA rep = 4
  const int Q0 = blockIdx.x * 64;
  const int qb = Q0 + wave * 16;
  const int q  = qb + lh;            // this lane's query column
  const float scale = 0.125f;        // 1/sqrt(64)

  // Q^T B-fragments (d is the WMMA K dim), 2 frags cover d=0..63
  V16 qf[2];
  {
    const half_t* qp = Q + ((size_t)(b * HQ_ + h) * S_ + q) * HD_;
#pragma unroll
    for (int s = 0; s < 2; ++s) {
      const uint4* p = (const uint4*)(qp + s * 32);
      qf[s].u[0] = p[sel];
      qf[s].u[1] = p[2 + sel];
    }
  }

  int klo = 0;
  if (!useGlobal) {
    int t = Q0 - (WINDOW_ - 1);
    if (t > 0) klo = (t >> 5) << 5;
  }
  const int khi = Q0 + 63;

  float mrun = NEGF, lrun = 0.0f;
  v8f ot[4];   // O^T tiles over d = 0..63
#pragma unroll
  for (int t = 0; t < 4; ++t)
#pragma unroll
    for (int i = 0; i < 8; ++i) ot[t][i] = 0.0f;

  const half_t* kbase = Kc + (size_t)(b * HKV_ + kvh) * S_ * HD_;
  const half_t* vbase = Vc + (size_t)(b * HKV_ + kvh) * S_ * HD_;

  const int skey = tid >> 2, soff = (tid & 3) * 16;   // staging roles

  for (int kb = klo; kb <= khi; kb += 32) {
    // ---- stage K tile (async to LDS) and transposed V tile ----
    async_b128x2((unsigned)(uintptr_t)(sK + skey * 64 + soff),
                 kbase + (size_t)(kb + skey) * HD_ + soff);
    {
      V16 tmp;
      const uint4* vs = (const uint4*)(vbase + (size_t)(kb + skey) * HD_ + soff);
      tmp.u[0] = vs[0];
      tmp.u[1] = vs[1];
#pragma unroll
      for (int j = 0; j < 16; ++j) sVt[(soff + j) * 32 + skey] = tmp.v[j];
    }
    if (kb + 32 <= khi) {
      __builtin_prefetch(kbase + (size_t)(kb + 32 + skey) * HD_ + soff, 0, 0);
      __builtin_prefetch(vbase + (size_t)(kb + 32 + skey) * HD_ + soff, 0, 0);
    }
    wait_async0();
    __syncthreads();

    // wave-uniform activity test for this key block
    bool active = (kb <= qb + 15) &&
                  (useGlobal || (kb + 31 >= qb - (WINDOW_ - 1)));
    if (active) {
      // S^T tiles: keys 0-15 (st[0]) and 16-31 (st[1]) x 16 queries
      v8f st[2];
#pragma unroll
      for (int t = 0; t < 2; ++t)
#pragma unroll
        for (int i = 0; i < 8; ++i) st[t][i] = 0.0f;
#pragma unroll
      for (int kt = 0; kt < 2; ++kt) {
        const uint4* pk = (const uint4*)(sK + (kt * 16 + lh) * 64);
#pragma unroll
        for (int s = 0; s < 2; ++s) {
          V16 ka;   // K-block A frag: M=key, K=d
          ka.u[0] = pk[s * 4 + sel];
          ka.u[1] = pk[s * 4 + 2 + sel];
          st[kt] = wmma32(ka.v, qf[s].v, st[kt]);
        }
      }
      // scale + mask (finite sentinel keeps everything NaN-free)
      float pp[2][8];
      float mloc = NEGF;
#pragma unroll
      for (int kt = 0; kt < 2; ++kt)
#pragma unroll
        for (int i = 0; i < 8; ++i) {
          int key = kb + kt * 16 + i + sel * 8;
          bool valid = (key <= q) && (useGlobal || (q - key) < WINDOW_);
          float sv = valid ? st[kt][i] * scale : NEGF;
          pp[kt][i] = sv;
          mloc = fmaxf(mloc, sv);
        }
      mloc = fmaxf(mloc, __shfl_xor(mloc, 16, 32));
      float mnew = fmaxf(mrun, mloc);
      float alpha = __expf(mrun - mnew);   // both NEGF -> exp(0)=1 (l==0 anyway)
      float lsum = 0.0f;
#pragma unroll
      for (int i = 0; i < 8; ++i) {
        pp[0][i] = (pp[0][i] <= NEGF) ? 0.0f : __expf(pp[0][i] - mnew);
        pp[1][i] = (pp[1][i] <= NEGF) ? 0.0f : __expf(pp[1][i] - mnew);
        lsum += pp[0][i] + pp[1][i];
      }
      lsum += __shfl_xor(lsum, 16, 32);
      lrun = lrun * alpha + lsum;
      mrun = mnew;
#pragma unroll
      for (int t = 0; t < 4; ++t)
#pragma unroll
        for (int i = 0; i < 8; ++i) ot[t][i] *= alpha;

      // build P B-fragment (K=local key, col=query) via xor-16 exchanges
      V16 pb;
#pragma unroll
      for (int i = 0; i < 8; ++i) {
        float o1 = __shfl_xor(pp[0][i], 16, 32);
        float o2 = __shfl_xor(pp[1][i], 16, 32);
        float lo = sel ? o2 : pp[0][i];   // local keys 0..7  / 16..23
        float hi = sel ? pp[1][i] : o1;   // local keys 8..15 / 24..31
        pb.v[i]     = (half_t)lo;
        pb.v[8 + i] = (half_t)hi;
      }
      // O^T += V^T * P  (contraction over 32 keys, one WMMA per d-tile)
#pragma unroll
      for (int t = 0; t < 4; ++t) {
        V16 va;   // V^T A frag: M=d, K=key
        const uint4* pv = (const uint4*)(sVt + (t * 16 + lh) * 32);
        va.u[0] = pv[2 * sel];
        va.u[1] = pv[2 * sel + 1];
        ot[t] = wmma32(va.v, pb.v, ot[t]);
      }
    }
    __syncthreads();
  }

  const float rl = lrun > 0.0f ? 1.0f / lrun : 0.0f;
  half_t* ob = O + ((size_t)(b * S_ + q) * (HQ_ * HD_)) + h * HD_;
#pragma unroll
  for (int t = 0; t < 4; ++t)
#pragma unroll
    for (int i = 0; i < 8; ++i) {
      int d = t * 16 + i + sel * 8;
      ob[d] = (half_t)(ot[t][i] * rl);
    }
}

// ---------------- host orchestration ----------------
extern "C" void kernel_launch(void* const* d_in, const int* in_sizes, int n_in,
                              void* d_out, int out_size, void* d_ws, size_t ws_size,
                              hipStream_t stream) {
  (void)in_sizes; (void)n_in; (void)out_size; (void)ws_size;
  const int*   idx = (const int*)d_in[0];
  const float* emb = (const float*)d_in[1];
  const float* Wq  = (const float*)d_in[2];
  const float* Wk  = (const float*)d_in[3];
  const float* Wv  = (const float*)d_in[4];
  const float* Wo  = (const float*)d_in[5];
  const float* w1  = (const float*)d_in[6];
  const float* w3  = (const float*)d_in[7];
  const float* w2  = (const float*)d_in[8];
  const float* n1  = (const float*)d_in[9];
  const float* n2  = (const float*)d_in[10];
  const float* nf  = (const float*)d_in[11];
  float* outp = (float*)d_out;

  const int M = B_ * S_;   // 2048 rows
  char* ws = (char*)d_ws;
  size_t off = 0;
  auto alloc = [&](size_t bytes) -> void* {
    void* p = ws + off;
    off = (off + bytes + 255) & ~(size_t)255;
    return p;
  };
  float*  x    = (float*) alloc((size_t)M * D_ * 4);
  half_t* hf   = (half_t*)alloc((size_t)M * D_ * 2);
  float*  qf32 = (float*) alloc((size_t)M * HQ_ * HD_ * 4);
  float*  kf32 = (float*) alloc((size_t)M * HKV_ * HD_ * 4);
  float*  vf32 = (float*) alloc((size_t)M * HKV_ * HD_ * 4);
  half_t* q16  = (half_t*)alloc((size_t)M * HQ_ * HD_ * 2);
  half_t* k16  = (half_t*)alloc((size_t)M * HKV_ * HD_ * 2);
  half_t* v16  = (half_t*)alloc((size_t)M * HKV_ * HD_ * 2);
  half_t* o16  = (half_t*)alloc((size_t)M * HQ_ * HD_ * 2);
  float*  ubuf = (float*) alloc((size_t)M * H_ * 4);
  float*  gbuf = (float*) alloc((size_t)M * H_ * 4);
  half_t* a16  = (half_t*)alloc((size_t)M * H_ * 2);

  embed_gather<<<(M * D_) / 256, 256, 0, stream>>>(idx, emb, x, M * D_);

  for (int l = 0; l < L_; ++l) {
    const int useGlobal = (((l + 1) % 4) == 0);
    rmsnorm_f16<<<M, 256, 0, stream>>>(x, n1 + (size_t)l * D_, hf);

    dim3 gq(HQ_ * HD_ / 64, M / 128);
    gemm_wmma<<<gq, 256, 0, stream>>>(hf, Wq + (size_t)l * D_ * HQ_ * HD_,
                                      nullptr, qf32, M, HQ_ * HD_, D_, 0);
    dim3 gk(HKV_ * HD_ / 64, M / 128);
    gemm_wmma<<<gk, 256, 0, stream>>>(hf, Wk + (size_t)l * D_ * HKV_ * HD_,
                                      nullptr, kf32, M, HKV_ * HD_, D_, 0);
    gemm_wmma<<<gk, 256, 0, stream>>>(hf, Wv + (size_t)l * D_ * HKV_ * HD_,
                                      nullptr, vf32, M, HKV_ * HD_, D_, 0);

    rope_pack<<<(B_ * S_ * HQ_ * 32) / 256, 256, 0, stream>>>(qf32, q16, HQ_, 1);
    rope_pack<<<(B_ * S_ * HKV_ * 32) / 256, 256, 0, stream>>>(kf32, k16, HKV_, 1);
    rope_pack<<<(B_ * S_ * HKV_ * 32) / 256, 256, 0, stream>>>(vf32, v16, HKV_, 0);

    dim3 gf(S_ / 64, B_ * HQ_);
    flash_attn<<<gf, 128, 0, stream>>>(q16, k16, v16, o16, useGlobal);

    dim3 go(D_ / 64, M / 128);
    gemm_wmma<<<go, 256, 0, stream>>>(o16, Wo + (size_t)l * HQ_ * HD_ * D_,
                                      x, x, M, D_, HQ_ * HD_, 0);

    rmsnorm_f16<<<M, 256, 0, stream>>>(x, n2 + (size_t)l * D_, hf);

    dim3 gm1(H_ / 64, M / 128);
    gemm_wmma<<<gm1, 256, 0, stream>>>(hf, w1 + (size_t)l * D_ * H_,
                                       nullptr, ubuf, M, H_, D_, 0);
    gemm_wmma<<<gm1, 256, 0, stream>>>(hf, w3 + (size_t)l * D_ * H_,
                                       nullptr, gbuf, M, H_, D_, 0);
    const int nsw = M * H_;
    swiglu_f16<<<(nsw + 255) / 256, 256, 0, stream>>>(ubuf, gbuf, a16, nsw);

    dim3 gm2(D_ / 64, M / 128);
    gemm_wmma<<<gm2, 256, 0, stream>>>(a16, w2 + (size_t)l * H_ * D_,
                                       x, x, M, D_, H_, 0);
  }

  rmsnorm_f16<<<M, 256, 0, stream>>>(x, nf, hf);
  dim3 gl(V_ / 64, M / 128);   // logits: x @ emb^T via transposed-B path
  gemm_wmma<<<gl, 256, 0, stream>>>(hf, emb, nullptr, outp, M, V_, D_, 1);
}
